// MultiheadAttention_37125697307146
// MI455X (gfx1250) — matrix operand order repaired and checked by reading, not compiled, not run on hardware
//
#include <hip/hip_runtime.h>

#define DM 512   // d_model
#define SQ 1024  // seq len
#define BB 2     // batch
#define CH 32    // number of chunks ("heads")
#define HD 16    // head dim (NH in reference)

typedef __attribute__((ext_vector_type(2)))  float v2f;
typedef __attribute__((ext_vector_type(4)))  float v4f;
typedef __attribute__((ext_vector_type(8)))  float v8f;
typedef __attribute__((ext_vector_type(4)))  unsigned short v4us;
typedef __attribute__((ext_vector_type(8)))  unsigned short v8us;
typedef __attribute__((ext_vector_type(16))) unsigned short v16us;
typedef __attribute__((ext_vector_type(4)))  __bf16 v4bf;
typedef __attribute__((ext_vector_type(8)))  __bf16 v8bf;
typedef __attribute__((ext_vector_type(16))) __bf16 v16bf;

// ---------------------------------------------------------------------------
// Bulk f32 -> bf16 (RNE via fptrunc; backend may select v_cvt_pk_bf16_f32).
// ---------------------------------------------------------------------------
__global__ __launch_bounds__(256)
void cvt_bf16(const float* __restrict__ src, unsigned short* __restrict__ dst, int n4) {
  const int i = blockIdx.x * 256 + threadIdx.x;
  if (i < n4) {
    v4f v = *(const v4f*)(src + 4 * (size_t)i);
    v4bf b = __builtin_convertvector(v, v4bf);
    *(v4us*)(dst + 4 * (size_t)i) = __builtin_bit_cast(v4us, b);
  }
}

// ---------------------------------------------------------------------------
// GEMM (NT), all-bf16 operands: Out[m,n] = sum_k X[m,k] * W[n,k]
// M=2048, N=512, K=512. One wave per 16x16 tile, 4 waves/block along N.
// mode 0: f32 row-major out.  mode 1: bf16 V^T-per-head out [c][b][h][s].
//
// bf16 A-frag (16x32 MxK, contiguous K, as B): lane l (row M=l&15) holds
//   K = [k0+16*hi .. +15] (elems 0..15), two 16B loads.
// bf16 B-frag (32x16 KxN): lane l = column N=l&15, 16 contiguous K values
//   starting at k0 (lanes 0-15) or k0+16 (lanes 16-31)  -> one 32B load.
// ---------------------------------------------------------------------------
__global__ __launch_bounds__(128)
void gemm_nt_bf16(const unsigned short* __restrict__ X,
                  const unsigned short* __restrict__ W,
                  float* __restrict__ outF, unsigned short* __restrict__ outVT,
                  int mode) {
  const int lane = threadIdx.x & 31;
  const int wave = threadIdx.x >> 5;
  const int hi  = lane >> 4;
  const int l15 = lane & 15;
  const int m0 = blockIdx.x * 16;
  const int n0 = (blockIdx.y * 4 + wave) * 16;

  const unsigned short* arow = X + (size_t)(m0 + l15) * DM;
  const unsigned short* brow = W + (size_t)(n0 + l15) * DM;

  v8f acc = {0.f, 0.f, 0.f, 0.f, 0.f, 0.f, 0.f, 0.f};
#pragma unroll 4
  for (int k0 = 0; k0 < DM; k0 += 32) {
    if ((k0 & 127) == 0) {
      __builtin_prefetch(arow + k0 + 128, 0, 1);
      __builtin_prefetch(brow + k0 + 128, 0, 1);
    }
    const int ab = k0 + 16 * hi;
    v8us alo = *(const v8us*)(arow + ab);
    v8us ahi = *(const v8us*)(arow + ab + 8);
    v16us av = __builtin_shufflevector(alo, ahi,
        0,1,2,3,4,5,6,7,8,9,10,11,12,13,14,15);
    v16us bv = *(const v16us*)(brow + k0 + 16 * hi);
    acc = __builtin_amdgcn_wmma_f32_16x16x32_bf16(
        false, __builtin_bit_cast(v16bf, av),
        false, __builtin_bit_cast(v16bf, bv),
        (short)0, acc, false, false);
  }

  if (mode == 0) {
#pragma unroll
    for (int r = 0; r < 8; ++r)
      outF[(size_t)(m0 + r + 8 * hi) * DM + n0 + l15] = acc[r];
  } else {
    v8bf ab = __builtin_convertvector(acc, v8bf);
    v8us au = __builtin_bit_cast(v8us, ab);
#pragma unroll
    for (int r = 0; r < 8; ++r) {
      const int m = m0 + r + 8 * hi;           // global row = b*S + s
      const int b = m >> 10, s = m & (SQ - 1);
      const int d = n0 + l15;                   // output feature
      const int c = d >> 4, h = d & 15;
      outVT[(size_t)(((c * BB + b) * HD) + h) * SQ + s] = au[r];
    }
  }
}

// ---------------------------------------------------------------------------
// Attention: one wave per (c, b, 16-row tile). Scores via exact f32 WMMA
// 16x16x4 (head dim 16 -> 4 K-steps), f32 softmax through a 64KB LDS strip,
// weights written f32 (float4 stores) to d_out, weights@V via bf16 WMMA,
// att scattered into the torch-faithful permuted out_pre (bf16).
// ---------------------------------------------------------------------------
__global__ __launch_bounds__(32)
void attn_kernel(const float* __restrict__ qf, const float* __restrict__ kf,
                 const unsigned short* __restrict__ vT,
                 unsigned short* __restrict__ outPre,
                 float* __restrict__ wOut) {
  extern __shared__ float sm[];
  float* sc     = sm;             // [16][1024] scores -> exp values
  float* redmax = sm + 16 * SQ;   // [16] row max
  float* invs   = redmax + 16;    // [16] 1/rowsum

  const int lane = threadIdx.x;
  const int hi  = lane >> 4;
  const int l15 = lane & 15;
  const int m0 = blockIdx.x * 16;
  const int b  = blockIdx.y;
  const int c  = blockIdx.z;

  // f32 16x4 A-frag: lane = row m0+l15, K = {kk0,kk0+1} (lo) / {kk0+2,kk0+3} (hi)
  const float* qrow = qf + (size_t)(b * SQ + m0 + l15) * DM + c * HD;
  v2f aq[4];
#pragma unroll
  for (int t = 0; t < 4; ++t) aq[t] = *(const v2f*)(qrow + 4 * t + 2 * hi);

  const float* kbase = kf + (size_t)(b * SQ) * DM + c * HD;

  float rmax[8];
#pragma unroll
  for (int r = 0; r < 8; ++r) rmax[r] = -3.4e38f;

  // scores = q kh^T / 4 ; store to LDS, track row max
  for (int n0 = 0; n0 < SQ; n0 += 16) {
    const float* krow = kbase + (size_t)(n0 + l15) * DM;
    v8f acc = {0.f, 0.f, 0.f, 0.f, 0.f, 0.f, 0.f, 0.f};
#pragma unroll
    for (int t = 0; t < 4; ++t) {
      v2f bf = *(const v2f*)(krow + 4 * t + 2 * hi);
      acc = __builtin_amdgcn_wmma_f32_16x16x4_f32(false, aq[t], false, bf,
                                                  (short)0, acc, false, false);
    }
#pragma unroll
    for (int r = 0; r < 8; ++r) {
      const float s = acc[r] * 0.25f;            // 1/sqrt(NH), NH=16
      sc[(r + 8 * hi) * SQ + n0 + l15] = s;
      rmax[r] = fmaxf(rmax[r], s);
    }
  }
  // row max: xor-shuffle reduce within each 16-lane half
#pragma unroll
  for (int r = 0; r < 8; ++r) {
#pragma unroll
    for (int off = 1; off < 16; off <<= 1)
      rmax[r] = fmaxf(rmax[r], __shfl_xor(rmax[r], off, 32));
  }
  if (l15 == 0) {
#pragma unroll
    for (int r = 0; r < 8; ++r) redmax[r + 8 * hi] = rmax[r];
  }
  __syncthreads();

  // exp + row sums: 2 lanes per row, 512 cols each, float4 strides
  {
    const int row = lane >> 1, half = lane & 1;
    const float mx = redmax[row];
    float psum = 0.f;
    float* p = sc + row * SQ + half * 512;
#pragma unroll 4
    for (int i = 0; i < 128; ++i) {
      v4f v = *(v4f*)(p + 4 * i);
      v4f e;
#pragma unroll
      for (int j = 0; j < 4; ++j) { e[j] = __expf(v[j] - mx); psum += e[j]; }
      *(v4f*)(p + 4 * i) = e;
    }
    psum += __shfl_xor(psum, 1, 32);
    if (half == 0) invs[row] = 1.0f / psum;
  }
  __syncthreads();

  // weights write (f32, float4) + att = weights @ vh (bf16 WMMA, K = 1024)
  const unsigned short* vrow = vT + (size_t)((c * BB + b) * HD + l15) * SQ;
  float* wrow = wOut + ((size_t)(c * BB + b) * SQ + m0 + l15) * SQ;
  const float inva = invs[l15];

  v8f att = {0.f, 0.f, 0.f, 0.f, 0.f, 0.f, 0.f, 0.f};
  for (int n0 = 0; n0 < SQ; n0 += 32) {
    const int kb = n0 + 8 * hi;
    const float* sp = sc + l15 * SQ + kb;
    v4f a0 = *(const v4f*)(sp)      * inva;
    v4f a1 = *(const v4f*)(sp + 4)  * inva;
    v4f a2 = *(const v4f*)(sp + 16) * inva;
    v4f a3 = *(const v4f*)(sp + 20) * inva;
    // normalized weights out (bandwidth-critical: 128-bit stores)
    *(v4f*)(wrow + kb)      = a0;
    *(v4f*)(wrow + kb + 4)  = a1;
    *(v4f*)(wrow + kb + 16) = a2;
    *(v4f*)(wrow + kb + 20) = a3;
    // bf16 A-frag (elems 0..7 = K kb..kb+7, elems 8..15 = K kb+16..kb+23)
    v8f lo8 = __builtin_shufflevector(a0, a1, 0,1,2,3,4,5,6,7);
    v8f hi8 = __builtin_shufflevector(a2, a3, 0,1,2,3,4,5,6,7);
    v8bf lob = __builtin_convertvector(lo8, v8bf);
    v8bf hib = __builtin_convertvector(hi8, v8bf);
    v16bf A = __builtin_shufflevector(lob, hib,
        0,1,2,3,4,5,6,7,8,9,10,11,12,13,14,15);
    v8us blo = *(const v8us*)(vrow + kb);
    v8us bhi = *(const v8us*)(vrow + kb + 16);
    v16us bu = __builtin_shufflevector(blo, bhi,
        0,1,2,3,4,5,6,7,8,9,10,11,12,13,14,15);
    att = __builtin_amdgcn_wmma_f32_16x16x32_bf16(
        false, A, false, __builtin_bit_cast(v16bf, bu),
        (short)0, att, false, false);
  }

  // scatter att tile into permuted out_pre (bf16), faithful to
  // att.transpose(0,2,1,3).reshape(B,-1,D):
  //   b' = c>>4 ; s' = (c&15)*64 + (s>>4) ; d' = (s&15)*32 + b*16 + h
  v8bf attb = __builtin_convertvector(att, v8bf);
  v8us attu = __builtin_bit_cast(v8us, attb);
#pragma unroll
  for (int r = 0; r < 8; ++r) {
    const int s = m0 + r + 8 * hi;
    const int h = l15;
    const int dp = (s & 15) * 32 + b * HD + h;
    const int sp2 = (c & 15) * 64 + (s >> 4);
    const int bp = c >> 4;
    outPre[(size_t)(bp * SQ + sp2) * DM + dp] = attu[r];
  }
}

extern "C" void kernel_launch(void* const* d_in, const int* in_sizes, int n_in,
                              void* d_out, int out_size, void* d_ws, size_t ws_size,
                              hipStream_t stream) {
  const float* Q  = (const float*)d_in[0];
  const float* K  = (const float*)d_in[1];
  const float* V  = (const float*)d_in[2];
  const float* Wq = (const float*)d_in[3];
  const float* Wk = (const float*)d_in[4];
  const float* Wv = (const float*)d_in[5];
  const float* Wo = (const float*)d_in[6];

  float* out     = (float*)d_out;                      // (2,1024,512)
  float* weights = out + (size_t)BB * SQ * DM;         // (32,2,1024,1024)

  char* ws = (char*)d_ws;
  float*          qf  = (float*)ws;                                 // 4 MB
  float*          kf  = (float*)(ws + ( 4u << 20));                 // 4 MB
  unsigned short* vT  = (unsigned short*)(ws + ( 8u << 20));        // 2 MB
  unsigned short* oP  = (unsigned short*)(ws + (10u << 20));        // 2 MB
  unsigned short* Qb  = (unsigned short*)(ws + (12u << 20));        // 2 MB
  unsigned short* Kb  = (unsigned short*)(ws + (14u << 20));        // 2 MB
  unsigned short* Vb  = (unsigned short*)(ws + (16u << 20));        // 2 MB
  unsigned short* Wqb = (unsigned short*)(ws + (18u << 20));        // 512 KB
  unsigned short* Wkb = (unsigned short*)(ws + (18u << 20) + (512u << 10));
  unsigned short* Wvb = (unsigned short*)(ws + (19u << 20));
  unsigned short* Wob = (unsigned short*)(ws + (19u << 20) + (512u << 10));

  // one-time bf16 conversion of inputs and weights
  const int nX4 = (BB * SQ * DM) / 4;   // 262144
  const int nW4 = (DM * DM) / 4;        // 65536
  cvt_bf16<<<nX4 / 256, 256, 0, stream>>>(Q,  Qb,  nX4);
  cvt_bf16<<<nX4 / 256, 256, 0, stream>>>(K,  Kb,  nX4);
  cvt_bf16<<<nX4 / 256, 256, 0, stream>>>(V,  Vb,  nX4);
  cvt_bf16<<<nW4 / 256, 256, 0, stream>>>(Wq, Wqb, nW4);
  cvt_bf16<<<nW4 / 256, 256, 0, stream>>>(Wk, Wkb, nW4);
  cvt_bf16<<<nW4 / 256, 256, 0, stream>>>(Wv, Wvb, nW4);
  cvt_bf16<<<nW4 / 256, 256, 0, stream>>>(Wo, Wob, nW4);

  // Projections: M=2048, N=512 -> (128 m-tiles) x (8 blocks * 4 waves)
  dim3 ggrid(128, 8);
  gemm_nt_bf16<<<ggrid, 128, 0, stream>>>(Qb, Wqb, qf, nullptr, 0);
  gemm_nt_bf16<<<ggrid, 128, 0, stream>>>(Kb, Wkb, kf, nullptr, 0);
  gemm_nt_bf16<<<ggrid, 128, 0, stream>>>(Vb, Wvb, nullptr, vT, 1);

  // Attention: (64 row-tiles) x (B=2) x (C=32), one wave each, 64KB LDS strip
  dim3 agrid(SQ / 16, BB, CH);
  size_t shmem = (size_t)(16 * SQ + 32) * sizeof(float);
  attn_kernel<<<agrid, 32, shmem, stream>>>(qf, kf, vT, oP, weights);

  // Output projection into d_out
  gemm_nt_bf16<<<ggrid, 128, 0, stream>>>(oP, Wob, out, nullptr, 0);
}
